// Attention_5325759447698
// MI455X (gfx1250) — matrix operand order repaired and checked by reading, not compiled, not run on hardware
//
#include <hip/hip_runtime.h>
#include <stdint.h>

// ---------------------------------------------------------------------------
// MHA forward for gfx1250 (MI455X), bf16 WMMA (v_wmma_f32_16x16x32_bf16),
// flash-attention with double-buffered async global->LDS tile staging
// (ASYNCcnt path), pre-packed weight B-fragments, batched fragment loads so
// WMMAs drain against partial loadcnt waits.
// ---------------------------------------------------------------------------

typedef __attribute__((ext_vector_type(16))) __bf16   v16bf;
typedef __attribute__((ext_vector_type(8)))  float    v8f;
typedef __attribute__((ext_vector_type(4)))  uint32_t u32x4;

union AB {
    v16bf    v;      // WMMA A or B fragment (16 bf16 per lane)
    u32x4    q[2];   // two 16B chunks
    uint32_t w[8];   // dword view (one packed bf16 pair each)
};

static __device__ inline v8f wmma_bf16(v16bf a, v16bf b, v8f c) {
    // D = A(16x32 bf16) * B(32x16 bf16) + C(16x16 f32)
    return __builtin_amdgcn_wmma_f32_16x16x32_bf16(
        /*neg_a=*/false, a, /*neg_b=*/false, b,
        /*c_mod=*/(short)0, c, /*reuse_a=*/false, /*reuse_b=*/false);
}

// Low 32 bits of a flat shared-memory address == LDS byte offset (ISA: LDS
// aperture places the LDS address in addr[31:0]).
static __device__ inline uint32_t lds_off(const void* p) {
    return (uint32_t)(uintptr_t)p;
}

// Async DMA: 16B per lane, global -> LDS, tracked by ASYNCcnt.
static __device__ inline void async_copy_b128(uint32_t lds_dst, const void* gsrc) {
    asm volatile("global_load_async_to_lds_b128 %0, %1, off"
                 :: "v"(lds_dst), "v"((uint64_t)(uintptr_t)gsrc) : "memory");
}
static __device__ inline void wait_async() {
    asm volatile("s_wait_asynccnt 0x0" ::: "memory");
}

// ---------------------------------------------------------------------------
// Kernel 1: attn_input = x + pos_embed (bf16), xv = x (bf16)
// ---------------------------------------------------------------------------
__global__ void k_prep(const float* __restrict__ x, const float* __restrict__ pe,
                       __bf16* __restrict__ ain, __bf16* __restrict__ xv, int n) {
    int i = blockIdx.x * blockDim.x + threadIdx.x;
    if (i < n) {
        float xi = x[i];
        ain[i] = (__bf16)(xi + pe[i]);
        xv[i]  = (__bf16)xi;
    }
}

// ---------------------------------------------------------------------------
// Kernel 2: pack a row-major f32 [K][N] matrix (nMat of them) into WMMA
// B-fragment order: dst[(mat*K/32*N/16 + kk*N/16 + nt)*256 + lane*8 + r]
// B 32x16 bf16 layout: lane holds column n = nt*16 + lane%16; dword r packs
// k-pair (k0, k0+1) with k0 = 2*(r&3) + 8*(lane/16) + 16*(r/4).
// ---------------------------------------------------------------------------
__global__ void k_packB(const float* __restrict__ src, uint32_t* __restrict__ dst,
                        int K, int N, int nMat) {
    int w    = (int)((blockIdx.x * blockDim.x + threadIdx.x) >> 5);
    int lane = threadIdx.x & 31;
    int fragsPerMat = (K / 32) * (N / 16);
    if (w >= nMat * fragsPerMat) return;
    int mat = w / fragsPerMat;
    int f   = w % fragsPerMat;
    int kk  = f / (N / 16);
    int nt  = f % (N / 16);
    int hlf = lane >> 4;
    int n   = nt * 16 + (lane & 15);
    const float* s = src + (size_t)mat * K * N;
    uint32_t* d = dst + (size_t)w * 256 + lane * 8;
#pragma unroll
    for (int r = 0; r < 8; ++r) {
        int k0 = kk * 32 + 2 * (r & 3) + 8 * hlf + 16 * (r >> 2);
        __bf16 lo = (__bf16)s[(size_t)k0 * N + n];
        __bf16 hi = (__bf16)s[(size_t)(k0 + 1) * N + n];
        d[r] = (uint32_t)__builtin_bit_cast(unsigned short, lo) |
               ((uint32_t)__builtin_bit_cast(unsigned short, hi) << 16);
    }
}

// ---------------------------------------------------------------------------
// Kernel 3: QKV projections. One wave per (b, head, 16-seq tile).
// C = A(16x1024) * W(1024x64); K-loop in 32-steps. Per matrix: batch-load 4
// B-fragments (one s_clause of 8 b128), then 4 WMMAs drain with partial
// loadcnt waits. Q,K stored bf16 [b*h][s][64]; V transposed [b*h][64][S].
// ---------------------------------------------------------------------------
__global__ void __launch_bounds__(256)
k_qkv(const __bf16* __restrict__ ain, const __bf16* __restrict__ xv,
      const uint32_t* __restrict__ wq, const uint32_t* __restrict__ wk,
      const uint32_t* __restrict__ wv,
      const float* __restrict__ bq, const float* __restrict__ bk,
      const float* __restrict__ bv,
      __bf16* __restrict__ Qo, __bf16* __restrict__ Ko, __bf16* __restrict__ Vo) {
    const int lane = threadIdx.x & 31;
    const int w    = (int)((blockIdx.x * blockDim.x + threadIdx.x) >> 5); // 0..4095
    const int qt  = w & 127;
    const int h   = (w >> 7) & 15;
    const int b   = w >> 11;
    const int hlf = lane >> 4;
    const int ln  = lane & 15;
    const int s0  = qt * 16;

    const v8f vzero = {0.f, 0.f, 0.f, 0.f, 0.f, 0.f, 0.f, 0.f};
    v8f accQ[4], accK[4], accV[4];
#pragma unroll
    for (int i = 0; i < 4; ++i) { accQ[i] = vzero; accK[i] = vzero; accV[i] = vzero; }

    // A-fragment row for this lane (A 16x32 bf16: row m = lane%16; two
    // contiguous 8-elem chunks at K = 8*hlf and K = 16 + 8*hlf).
    const __bf16* arow_qk = ain + ((size_t)b * 2048 + s0 + ln) * 1024;
    const __bf16* arow_v  = xv  + ((size_t)b * 2048 + s0 + ln) * 1024;

    for (int kk = 0; kk < 32; ++kk) {
        AB aqk, av;
        const int c1 = kk * 32 + 8 * hlf;
        aqk.q[0] = *(const u32x4*)(arow_qk + c1);
        aqk.q[1] = *(const u32x4*)(arow_qk + c1 + 16);
        av.q[0]  = *(const u32x4*)(arow_v + c1);
        av.q[1]  = *(const u32x4*)(arow_v + c1 + 16);

        const size_t fbase = ((size_t)h * 128 + kk * 4) * 256 + lane * 8;
        AB f[4];
        // ---- Q: batch-load 4 fragments, then 4 WMMAs ----
#pragma unroll
        for (int nt = 0; nt < 4; ++nt) {
            const size_t fi = fbase + (size_t)nt * 256;
            f[nt].q[0] = *(const u32x4*)(wq + fi);
            f[nt].q[1] = *(const u32x4*)(wq + fi + 4);
        }
#pragma unroll
        for (int nt = 0; nt < 4; ++nt) accQ[nt] = wmma_bf16(aqk.v, f[nt].v, accQ[nt]);
        // ---- K ----
#pragma unroll
        for (int nt = 0; nt < 4; ++nt) {
            const size_t fi = fbase + (size_t)nt * 256;
            f[nt].q[0] = *(const u32x4*)(wk + fi);
            f[nt].q[1] = *(const u32x4*)(wk + fi + 4);
        }
#pragma unroll
        for (int nt = 0; nt < 4; ++nt) accK[nt] = wmma_bf16(aqk.v, f[nt].v, accK[nt]);
        // ---- V ----
#pragma unroll
        for (int nt = 0; nt < 4; ++nt) {
            const size_t fi = fbase + (size_t)nt * 256;
            f[nt].q[0] = *(const u32x4*)(wv + fi);
            f[nt].q[1] = *(const u32x4*)(wv + fi + 4);
        }
#pragma unroll
        for (int nt = 0; nt < 4; ++nt) accV[nt] = wmma_bf16(av.v, f[nt].v, accV[nt]);
    }

    // Epilogue: C layout m = r + 8*hlf, n = lane%16. Add bias, store.
    const size_t bh = (size_t)b * 16 + h;
#pragma unroll
    for (int nt = 0; nt < 4; ++nt) {
        const int n = nt * 16 + ln;
        const float aQ = bq[h * 64 + n], aK = bk[h * 64 + n], aV = bv[h * 64 + n];
#pragma unroll
        for (int r = 0; r < 8; ++r) {
            const int m = r + 8 * hlf;
            const size_t row = bh * 2048 + s0 + m;
            Qo[row * 64 + n] = (__bf16)(accQ[nt][r] + aQ);
            Ko[row * 64 + n] = (__bf16)(accK[nt][r] + aK);
            Vo[(bh * 64 + n) * 2048 + s0 + m] = (__bf16)(accV[nt][r] + aV);
        }
    }
}

// ---------------------------------------------------------------------------
// Kernel 4: causal flash attention. One block per (b, head, 8 query tiles);
// one wave per 16-query tile. 32-key K/V tiles are double-buffered in LDS:
// async DMA for tile kb+1 is issued right after the barrier, overlapping the
// compute on tile kb. One barrier per iteration.
// ---------------------------------------------------------------------------
#define K_TILE_ELTS (32 * 72)   // 32 keys x 64 hd, row padded to 72
#define V_TILE_ELTS (64 * 40)   // 64 hd x 32 seq, row padded to 40
__global__ void __launch_bounds__(256)
k_attn(const __bf16* __restrict__ Q, const __bf16* __restrict__ K,
       const __bf16* __restrict__ Vt, __bf16* __restrict__ Z) {
    __shared__ __align__(16) __bf16 kt_s[2][K_TILE_ELTS];
    __shared__ __align__(16) __bf16 vt_s[2][V_TILE_ELTS];
    __shared__ __align__(16) __bf16 pt_s[8 * 16 * 40]; // per-wave P tiles

    const int tid  = threadIdx.x;
    const int lane = tid & 31;
    const int wl   = tid >> 5;
    const int blk  = blockIdx.x;    // 0..511
    const int qg   = blk & 15;      // group of 8 query tiles
    const int h    = (blk >> 4) & 15;
    const int b    = blk >> 8;
    const int qt   = qg * 8 + wl;   // this wave's query tile
    const int hlf  = lane >> 4;
    const int ln   = lane & 15;
    __bf16* pl = pt_s + wl * (16 * 40);

    const size_t bh = (size_t)b * 16 + h;
    const __bf16* Qb = Q  + bh * 2048 * 64;
    const __bf16* Kb = K  + bh * 2048 * 64;
    const __bf16* Vb = Vt + bh * 64 * 2048;

    // Q A-fragments for d_head chunks [0,32) and [32,64).
    AB aq0, aq1;
    {
        const __bf16* qrow = Qb + (size_t)(qt * 16 + ln) * 64 + 8 * hlf;
        aq0.q[0] = *(const u32x4*)(qrow);
        aq0.q[1] = *(const u32x4*)(qrow + 16);
        aq1.q[0] = *(const u32x4*)(qrow + 32);
        aq1.q[1] = *(const u32x4*)(qrow + 48);
    }

    const v8f vzero = {0.f, 0.f, 0.f, 0.f, 0.f, 0.f, 0.f, 0.f};
    v8f acc[4];
    float mrow[8], lrow[8];
#pragma unroll
    for (int i = 0; i < 4; ++i) acc[i] = vzero;
#pragma unroll
    for (int r = 0; r < 8; ++r) { mrow[r] = -1e30f; lrow[r] = 0.f; }

    // Cooperative staging assignment: 256 threads, one 16B async chunk per
    // thread per array. K tile rows: 128B data / 144B LDS stride. V tile
    // rows: 64B data / 80B LDS stride.
    const int krow = tid >> 3, kchunk = tid & 7;
    const int vrow = tid >> 2, vchunk = tid & 3;
    const __bf16* ksrc = Kb + (size_t)krow * 64 + kchunk * 8;
    const __bf16* vsrc = Vb + (size_t)vrow * 2048 + vchunk * 8;
    const uint32_t kdst0 = lds_off(kt_s) + (uint32_t)(krow * 144 + kchunk * 16);
    const uint32_t vdst0 = lds_off(vt_s) + (uint32_t)(vrow * 80 + vchunk * 16);

    const int kbmax_w = (qt * 16 + 15) >> 5;  // this wave's causal extent
    const int kbmax_b = 4 * qg + 3;           // block max (wave wl==7)

    // Prologue: stage tile 0 into buffer 0.
    async_copy_b128(kdst0, ksrc);
    async_copy_b128(vdst0, vsrc);

    for (int kb = 0; kb <= kbmax_b; ++kb) {
        const int cur = kb & 1;
        wait_async();        // current tile's DMA (only outstanding asyncs)
        __syncthreads();     // all waves' copies visible; prior reads of the
                             // alternate buffer (iter kb-1) are complete.
        if (kb < kbmax_b) {  // prefetch next tile into alternate buffer
            const int nxt = cur ^ 1;
            async_copy_b128(kdst0 + nxt * (K_TILE_ELTS * 2),
                            ksrc + (size_t)(kb + 1) * 32 * 64);
            async_copy_b128(vdst0 + nxt * (V_TILE_ELTS * 2),
                            vsrc + (size_t)(kb + 1) * 32);
        }

        if (kb <= kbmax_w) {
            const __bf16* ktile = kt_s[cur];
            const __bf16* vtile = vt_s[cur];
            v8f sc[2];
#pragma unroll
            for (int n16 = 0; n16 < 2; ++n16) {
                const int keyl = n16 * 16 + ln;        // key within tile
                const int key  = kb * 32 + keyl;       // global key
                const __bf16* krow_s = ktile + keyl * 72;
                AB bk0, bk1;
#pragma unroll
                for (int r = 0; r < 8; ++r) {
                    const int k0 = 2 * (r & 3) + 8 * hlf + 16 * (r >> 2);       // B-layout pair base
                    bk0.w[r] = *(const uint32_t*)(krow_s + k0);       // hd [0,32)
                    bk1.w[r] = *(const uint32_t*)(krow_s + 32 + k0);  // hd [32,64)
                }
                v8f c = vzero;
                c = wmma_bf16(aq0.v, bk0.v, c);
                c = wmma_bf16(aq1.v, bk1.v, c);
#pragma unroll
                for (int r = 0; r < 8; ++r) c[r] *= 0.125f;   // 1/sqrt(64)
                if (kb * 32 + 31 > qt * 16) {                 // causal mask region
#pragma unroll
                    for (int r = 0; r < 8; ++r) {
                        const int query = qt * 16 + r + 8 * hlf;
                        if (key > query) c[r] = -1e30f;
                    }
                }
                sc[n16] = c;
            }

            // Row max across the 16 lanes holding each row.
            float rm[8], rs[8];
#pragma unroll
            for (int r = 0; r < 8; ++r) rm[r] = fmaxf(sc[0][r], sc[1][r]);
            for (int off = 1; off < 16; off <<= 1) {
#pragma unroll
                for (int r = 0; r < 8; ++r) rm[r] = fmaxf(rm[r], __shfl_xor(rm[r], off, 32));
            }
#pragma unroll
            for (int r = 0; r < 8; ++r) {
                const float mn    = fmaxf(mrow[r], rm[r]);
                const float alpha = __expf(mrow[r] - mn);
                mrow[r] = mn;
                const float p0 = __expf(sc[0][r] - mn);
                const float p1 = __expf(sc[1][r] - mn);
                rs[r] = p0 + p1;
                lrow[r] *= alpha;
                acc[0][r] *= alpha; acc[1][r] *= alpha;
                acc[2][r] *= alpha; acc[3][r] *= alpha;
                const int m = r + 8 * hlf;                   // C layout row
                pl[m * 40 + ln]      = (__bf16)p0;
                pl[m * 40 + 16 + ln] = (__bf16)p1;
            }
            for (int off = 1; off < 16; off <<= 1) {
#pragma unroll
                for (int r = 0; r < 8; ++r) rs[r] += __shfl_xor(rs[r], off, 32);
            }
#pragma unroll
            for (int r = 0; r < 8; ++r) lrow[r] += rs[r];

            // P (C layout) -> A layout via per-wave LDS round trip.
            asm volatile("s_wait_dscnt 0x0" ::: "memory");
            AB pa;
            const __bf16* prow = pl + ln * 40 + 8 * hlf;
            pa.q[0] = *(const u32x4*)(prow);
            pa.q[1] = *(const u32x4*)(prow + 16);

#pragma unroll
            for (int nt = 0; nt < 4; ++nt) {
                AB bvf; // B = V[key][hd]; pairs contiguous in seq in vtile rows
                const __bf16* vrow_s = vtile + (nt * 16 + ln) * 40;
#pragma unroll
                for (int r = 0; r < 8; ++r) {
                    const int k0 = 2 * (r & 3) + 8 * hlf + 16 * (r >> 2);
                    bvf.w[r] = *(const uint32_t*)(vrow_s + k0);
                }
                acc[nt] = wmma_bf16(pa.v, bvf.v, acc[nt]);
            }
        }
        // No trailing barrier: next iteration's wait_async + barrier protect
        // the buffers (each wave's LDS reads are consumed before it signals).
    }

    // Normalize and store Z bf16 [b][s][head*64 + hd].
#pragma unroll
    for (int r = 0; r < 8; ++r) lrow[r] = 1.0f / lrow[r];
#pragma unroll
    for (int nt = 0; nt < 4; ++nt) {
#pragma unroll
        for (int r = 0; r < 8; ++r) {
            const int m = r + 8 * hlf;
            Z[((size_t)b * 2048 + qt * 16 + m) * 1024 + h * 64 + nt * 16 + ln] =
                (__bf16)(acc[nt][r] * lrow[r]);
        }
    }
}

// ---------------------------------------------------------------------------
// Kernel 5: out = Z (4096x1024) * W_O (1024x1024) + b_O, f32 output.
// One wave per 16x64 output tile; batch-load 4 B-fragments per k-step.
// ---------------------------------------------------------------------------
__global__ void __launch_bounds__(256)
k_out(const __bf16* __restrict__ Z, const uint32_t* __restrict__ woP,
      const float* __restrict__ bO, float* __restrict__ out) {
    const int lane = threadIdx.x & 31;
    const int w    = (int)((blockIdx.x * blockDim.x + threadIdx.x) >> 5); // 0..4095
    const int nt64 = w & 15;
    const int mt   = w >> 4;
    const int hlf  = lane >> 4;
    const int ln   = lane & 15;

    const v8f vzero = {0.f, 0.f, 0.f, 0.f, 0.f, 0.f, 0.f, 0.f};
    v8f acc[4];
#pragma unroll
    for (int i = 0; i < 4; ++i) acc[i] = vzero;

    const __bf16* arow = Z + ((size_t)mt * 16 + ln) * 1024;
    for (int kk = 0; kk < 32; ++kk) {
        AB a;
        const int c1 = kk * 32 + 8 * hlf;
        a.q[0] = *(const u32x4*)(arow + c1);
        a.q[1] = *(const u32x4*)(arow + c1 + 16);
        AB f[4];
#pragma unroll
        for (int j = 0; j < 4; ++j) {
            const size_t fi = ((size_t)kk * 64 + nt64 * 4 + j) * 256 + lane * 8;
            f[j].q[0] = *(const u32x4*)(woP + fi);
            f[j].q[1] = *(const u32x4*)(woP + fi + 4);
        }
#pragma unroll
        for (int j = 0; j < 4; ++j) acc[j] = wmma_bf16(a.v, f[j].v, acc[j]);
    }
#pragma unroll
    for (int j = 0; j < 4; ++j) {
        const int n = nt64 * 64 + j * 16 + ln;
        const float bo = bO[n];
#pragma unroll
        for (int r = 0; r < 8; ++r) {
            const int m = r + 8 * hlf;
            out[((size_t)mt * 16 + m) * 1024 + n] = acc[j][r] + bo;
        }
    }
}

// ---------------------------------------------------------------------------
extern "C" void kernel_launch(void* const* d_in, const int* in_sizes, int n_in,
                              void* d_out, int out_size, void* d_ws, size_t ws_size,
                              hipStream_t stream) {
    (void)in_sizes; (void)n_in; (void)out_size; (void)ws_size;
    const float* x  = (const float*)d_in[0];
    const float* pe = (const float*)d_in[1];
    const float* WQ = (const float*)d_in[2];
    const float* bQ = (const float*)d_in[3];
    const float* WK = (const float*)d_in[4];
    const float* bK = (const float*)d_in[5];
    const float* WV = (const float*)d_in[6];
    const float* bV = (const float*)d_in[7];
    const float* WO = (const float*)d_in[8];
    const float* bO = (const float*)d_in[9];
    float* out = (float*)d_out;

    char* p = (char*)d_ws;
    const size_t MB = 1024 * 1024;
    __bf16* AIN = (__bf16*)p; p += 8 * MB;   // [2][2048][1024] bf16
    __bf16* XV  = (__bf16*)p; p += 8 * MB;   // [2][2048][1024] bf16
    __bf16* Qb  = (__bf16*)p; p += 8 * MB;   // [32][2048][64]  bf16
    __bf16* Kb  = (__bf16*)p; p += 8 * MB;   // [32][2048][64]  bf16
    __bf16* Vt  = (__bf16*)p; p += 8 * MB;   // [32][64][2048]  bf16
    __bf16* Zb  = (__bf16*)p; p += 8 * MB;   // [2][2048][1024] bf16
    uint32_t* WqP = (uint32_t*)p; p += 2 * MB;
    uint32_t* WkP = (uint32_t*)p; p += 2 * MB;
    uint32_t* WvP = (uint32_t*)p; p += 2 * MB;
    uint32_t* WoP = (uint32_t*)p; p += 2 * MB;

    k_prep<<<16384, 256, 0, stream>>>(x, pe, AIN, XV, 2 * 2048 * 1024);
    k_packB<<<256, 256, 0, stream>>>(WQ, WqP, 1024, 64, 16);
    k_packB<<<256, 256, 0, stream>>>(WK, WkP, 1024, 64, 16);
    k_packB<<<256, 256, 0, stream>>>(WV, WvP, 1024, 64, 16);
    k_packB<<<256, 256, 0, stream>>>(WO, WoP, 1024, 1024, 1);
    k_qkv<<<512, 256, 0, stream>>>(AIN, XV, WqP, WkP, WvP, bQ, bK, bV, Qb, Kb, Vt);
    k_attn<<<512, 256, 0, stream>>>(Qb, Kb, Vt, Zb);
    k_out<<<512, 256, 0, stream>>>(Zb, WoP, bO, out);
}
